// LRSA_63471026700540
// MI455X (gfx1250) — matrix operand order repaired and checked by reading, not compiled, not run on hardware
//
#include <hip/hip_runtime.h>
#include <hip/hip_bf16.h>

// ---------------------------------------------------------------------------
// Problem constants (from reference): B=4, T=2048, DIM=512, H=8, HD=64
// ---------------------------------------------------------------------------
#define BB    4
#define TT    2048
#define DIMM  512
#define HH    8
#define HDD   64
#define ALPHA 100.0f
#define EPSS  1e-10f

typedef __attribute__((ext_vector_type(16))) _Float16 v16h;
typedef __attribute__((ext_vector_type(8)))  _Float16 v8h;
typedef __attribute__((ext_vector_type(8)))  float    v8f;

// D = A(16x32 f16) * B(32x16 f16) + C(16x16 f32)
static __device__ __forceinline__ v8f wmma_f16(v16h a, v16h b, v8f c) {
    return __builtin_amdgcn_wmma_f32_16x16x32_f16(
        /*neg_a=*/false, a, /*neg_b=*/false, b,
        /*c_mod=*/(short)0, c, /*reuse_a=*/false, /*reuse_b=*/false);
}

// CDNA5 async global->LDS DMA (one B128 per active lane), tracked by ASYNCcnt
static __device__ __forceinline__ void async_load_b128(unsigned lds_off,
                                                       const void* gaddr) {
    asm volatile("global_load_async_to_lds_b128 %0, %1, off"
                 :: "v"(lds_off), "v"(gaddr) : "memory");
}
static __device__ __forceinline__ void wait_asynccnt0() {
    asm volatile("s_wait_asynccnt 0x0" ::: "memory");
}

// ---------------------------------------------------------------------------
// K1: convert weights fp32 -> f16
// ---------------------------------------------------------------------------
__global__ void k_convert_w(const float* __restrict__ wqkv,
                            const float* __restrict__ wproj,
                            _Float16* __restrict__ wq16,
                            _Float16* __restrict__ wp16) {
    int i = blockIdx.x * blockDim.x + threadIdx.x;
    if (i < DIMM * 2 * DIMM) wq16[i] = (_Float16)wqkv[i];
    if (i < DIMM * DIMM)     wp16[i] = (_Float16)wproj[i];
}

// ---------------------------------------------------------------------------
// K2: per-row: x -> f16 copy, ||x[b,t]||^2, atomicMax into bmax_sq_bits[b]
// one wave (32 lanes) per row
// ---------------------------------------------------------------------------
__global__ void k_xrow(const float* __restrict__ x,
                       _Float16* __restrict__ x16,
                       unsigned* __restrict__ bmax_bits) {
    const int row  = blockIdx.x;          // 0 .. B*T-1
    const int lane = threadIdx.x;         // wave32
    const float*  xp = x   + (size_t)row * DIMM;
    _Float16*     op = x16 + (size_t)row * DIMM;
    float s = 0.f;
    for (int j = lane; j < DIMM; j += 32) {
        float v = xp[j];
        op[j] = (_Float16)v;
        s += v * v;
    }
    #pragma unroll
    for (int m = 16; m; m >>= 1) s += __shfl_xor(s, m, 32);
    if (lane == 0) {
        int b = row >> 11;                 // row / T
        atomicMax(&bmax_bits[b], __float_as_uint(s));  // s >= 0 -> bit-monotonic
    }
}

// ---------------------------------------------------------------------------
// K3: qkv GEMM: [8192,512]f16 @ [512,1024]f16 -> scatter to q/v [b,h,t,d] f16
// 256 threads = 8 waves; wave tile 16x16; block tile 64(M) x 32(N)
// ---------------------------------------------------------------------------
__global__ __launch_bounds__(256) void k_qkv_gemm(
        const _Float16* __restrict__ A, const _Float16* __restrict__ Bm,
        _Float16* __restrict__ qo, _Float16* __restrict__ vo) {
    const int lane = threadIdx.x & 31, w = threadIdx.x >> 5;
    const int mw = w & 3, nw = w >> 2;
    const int m0 = blockIdx.x * 64 + mw * 16;
    const int n0 = blockIdx.y * 32 + nw * 16;
    const int g = lane >> 4, ln = lane & 15;

    const _Float16* arow = A + (size_t)(m0 + ln) * DIMM;
    v8f acc = {};
    for (int kc = 0; kc < DIMM; kc += 32) {
        // A fragment: lane m=ln,
        //   elem j <- K = kc + 16g + j (contiguous, matches B)
        v16h af;
        const _Float16* ap = arow + kc + 16 * g;
        __builtin_prefetch(ap + 64, 0, 0);
        #pragma unroll
        for (int e = 0; e < 8; ++e) { af[e] = ap[e]; af[8 + e] = ap[8 + e]; }
        // B fragment: lane n=n0+ln, elem j <- K row kc+16g+j
        v16h bf;
        const _Float16* bp = Bm + (size_t)(kc + 16 * g) * (2 * DIMM) + n0 + ln;
        #pragma unroll
        for (int j = 0; j < 16; ++j) bf[j] = bp[(size_t)j * (2 * DIMM)];
        acc = wmma_f16(af, bf, acc);
    }
    // column c -> (d = c>>4, k = (c>>3)&1, h = c&7); rows i+8g
    const int c = n0 + ln;
    const int h = c & 7, kk = (c >> 3) & 1, d = c >> 4;
    _Float16* dst = (kk == 0) ? qo : vo;
    #pragma unroll
    for (int i = 0; i < 8; ++i) {
        int mr = m0 + 8 * g + i;
        int b = mr >> 11, t = mr & (TT - 1);
        dst[(((size_t)(b * HH + h) * TT) + t) * HDD + d] = (_Float16)acc[i];
    }
}

// ---------------------------------------------------------------------------
// K4: qsq[b,h,t] = sum_d q^2 ; a = sqrt(sum_t qsq); scale = ALPHA/(a*bmax+eps)
// one block per (b,h)
// ---------------------------------------------------------------------------
__global__ __launch_bounds__(256) void k_qsq_scale(
        const _Float16* __restrict__ q, float* __restrict__ qsq,
        const unsigned* __restrict__ bmax_bits, float* __restrict__ scalev) {
    const int bh = blockIdx.x, tid = threadIdx.x;
    __shared__ float red[256];
    float asum = 0.f;
    for (int t = tid; t < TT; t += 256) {
        const _Float16* row = q + ((size_t)bh * TT + t) * HDD;
        float s = 0.f;
        #pragma unroll
        for (int d = 0; d < HDD; ++d) { float v = (float)row[d]; s += v * v; }
        qsq[(size_t)bh * TT + t] = s;
        asum += s;
    }
    red[tid] = asum;
    __syncthreads();
    for (int st = 128; st > 0; st >>= 1) {
        if (tid < st) red[tid] += red[tid + st];
        __syncthreads();
    }
    if (tid == 0) {
        float a  = sqrtf(red[0]);
        float bm = sqrtf(__uint_as_float(bmax_bits[bh >> 3]));
        scalev[bh] = ALPHA / (a * bm + EPSS);
    }
}

// ---------------------------------------------------------------------------
// K5: flash attention. grid (T/128, B*H); 8 waves/block; wave = 16 query rows.
// Block-lockstep s-loop in tiles of 32:
//   K tile (32x64 f16 = 4KB) staged by async global->LDS DMA (1 B128/lane),
//   V tile staged transposed in LDS so P@V B-fragments are contiguous reads,
//   P transposed to an A-fragment through a per-wave LDS buffer.
// ---------------------------------------------------------------------------
__global__ __launch_bounds__(256) void k_attn(
        const _Float16* __restrict__ q, const _Float16* __restrict__ v,
        const float* __restrict__ qsq, const float* __restrict__ scalev,
        _Float16* __restrict__ oattn) {
    __shared__ _Float16 kls[32][64];      // K tile, natural [s][d]       4 KB
    __shared__ _Float16 vls[64][48];      // V tile, transposed [d][s]+pad 6 KB
    __shared__ _Float16 pbuf[8][16][32];  // per-wave P staging            8 KB

    const int tid  = threadIdx.x;
    const int lane = tid & 31, w = tid >> 5;
    const int bh = blockIdx.y, b = bh >> 3, h = bh & 7;
    const int t0 = blockIdx.x * 128 + w * 16;
    const int g = lane >> 4, ln = lane & 15;

    const float sc = scalev[bh];
    const _Float16* qb  = q   + (size_t)bh * TT * HDD;
    const _Float16* vb  = v   + (size_t)bh * TT * HDD;
    const float*    qsb = qsq + (size_t)bh * TT;

    // raw LDS byte offset of this thread's async-DMA destination in kls
    const unsigned k_lds_off =
        (unsigned)(size_t)(&kls[0][0]) + (unsigned)tid * 16u;

    // Q A-fragments for K-chunks d=[0,32) and d=[32,64)
    v16h aq0, aq1;
    {
        const _Float16* rp = qb + (size_t)(t0 + ln) * HDD;
        #pragma unroll
        for (int e = 0; e < 8; ++e) {
            aq0[e] = rp[16 * g + e];       aq0[8 + e] = rp[16 * g + 8 + e];
            aq1[e] = rp[32 + 16 * g + e];  aq1[8 + e] = rp[40 + 16 * g + e];
        }
    }
    float qm[8];
    #pragma unroll
    for (int i = 0; i < 8; ++i) qm[i] = qsb[t0 + 8 * g + i];

    float rowM[8], rowL[8];
    #pragma unroll
    for (int i = 0; i < 8; ++i) { rowM[i] = -3.0e38f; rowL[i] = 0.f; }
    v8f oacc[4] = {{}, {}, {}, {}};

    const int vr = tid >> 3, vc0 = (tid & 7) * 8;   // V staging coordinates

    for (int s0 = 0; s0 < TT; s0 += 32) {
        __syncthreads();   // previous iteration's consumers are done with LDS
        // ---- stage K tile: one async B128 per lane (tile is exactly 4KB) ----
        async_load_b128(k_lds_off, qb + (size_t)s0 * HDD + (size_t)tid * 8);
        // ---- stage V tile transposed: vls[d][s] ----
        {
            const v8h vv = *(const v8h*)(vb + (size_t)(s0 + vr) * HDD + vc0);
            #pragma unroll
            for (int e = 0; e < 8; ++e) vls[vc0 + e][vr] = vv[e];
        }
        wait_asynccnt0();
        __syncthreads();

        float st[2][8];
        #pragma unroll
        for (int ht = 0; ht < 2; ++ht) {
            const int scol = s0 + 16 * ht;
            // K^T B-fragments from the async-staged tile (contiguous LDS reads)
            const _Float16* kr = &kls[16 * ht + ln][0];
            v16h bk0 = *(const v16h*)(kr + 16 * g);        // d = 0..31 block
            v16h bk1 = *(const v16h*)(kr + 32 + 16 * g);   // d = 32..63 block
            v8f c = {};
            c = wmma_f16(aq0, bk0, c);
            c = wmma_f16(aq1, bk1, c);
            const float qsn = qsb[scol + ln];
            #pragma unroll
            for (int i = 0; i < 8; ++i)
                st[ht][i] = sc * (2.f * c[i] - qm[i] - qsn);
        }
        // online softmax over the 32 columns (row lives in a 16-lane group)
        #pragma unroll
        for (int i = 0; i < 8; ++i) {
            float t = fmaxf(st[0][i], st[1][i]);
            #pragma unroll
            for (int m = 8; m; m >>= 1) t = fmaxf(t, __shfl_xor(t, m, 16));
            const float mnew = fmaxf(rowM[i], t);
            const float al = __expf(rowM[i] - mnew);
            const float p0 = __expf(st[0][i] - mnew);
            const float p1 = __expf(st[1][i] - mnew);
            float rs = p0 + p1;
            #pragma unroll
            for (int m = 8; m; m >>= 1) rs += __shfl_xor(rs, m, 16);
            rowL[i] = rowL[i] * al + rs;
            rowM[i] = mnew;
            oacc[0][i] *= al; oacc[1][i] *= al; oacc[2][i] *= al; oacc[3][i] *= al;
            pbuf[w][8 * g + i][ln]      = (_Float16)p0;
            pbuf[w][8 * g + i][16 + ln] = (_Float16)p1;
        }
        // P as A-fragment (16x32 over s): same-wave LDS round-trip (in-order DS)
        v16h pa;
        {
            const v8h plo = *(const v8h*)&pbuf[w][ln][16 * g];
            const v8h phi = *(const v8h*)&pbuf[w][ln][16 * g + 8];
            #pragma unroll
            for (int e = 0; e < 8; ++e) { pa[e] = plo[e]; pa[8 + e] = phi[e]; }
        }
        // V B-fragments: contiguous reads from the transposed LDS tile
        #pragma unroll
        for (int nd = 0; nd < 4; ++nd) {
            v16h vbf = *(const v16h*)&vls[nd * 16 + ln][16 * g];
            oacc[nd] = wmma_f16(pa, vbf, oacc[nd]);
        }
    }
    // normalize and store to [b, t, h*64 + d] f16
    #pragma unroll
    for (int nd = 0; nd < 4; ++nd) {
        #pragma unroll
        for (int i = 0; i < 8; ++i) {
            float val = oacc[nd][i] / rowL[i];
            int r = t0 + 8 * g + i;
            oattn[((size_t)(b * TT + r)) * DIMM + h * HDD + nd * 16 + ln] =
                (_Float16)val;
        }
    }
}

// ---------------------------------------------------------------------------
// K6: projection GEMM: [8192,512]f16 @ [512,512]f16 + bias -> fp32 out
// ---------------------------------------------------------------------------
__global__ __launch_bounds__(256) void k_proj_gemm(
        const _Float16* __restrict__ A, const _Float16* __restrict__ Bm,
        const float* __restrict__ bias, float* __restrict__ out) {
    const int lane = threadIdx.x & 31, w = threadIdx.x >> 5;
    const int mw = w & 3, nw = w >> 2;
    const int m0 = blockIdx.x * 64 + mw * 16;
    const int n0 = blockIdx.y * 32 + nw * 16;
    const int g = lane >> 4, ln = lane & 15;

    const _Float16* arow = A + (size_t)(m0 + ln) * DIMM;
    v8f acc = {};
    for (int kc = 0; kc < DIMM; kc += 32) {
        v16h af;
        const _Float16* ap = arow + kc + 16 * g;
        __builtin_prefetch(ap + 64, 0, 0);
        #pragma unroll
        for (int e = 0; e < 8; ++e) { af[e] = ap[e]; af[8 + e] = ap[8 + e]; }
        v16h bf;
        const _Float16* bp = Bm + (size_t)(kc + 16 * g) * DIMM + n0 + ln;
        #pragma unroll
        for (int j = 0; j < 16; ++j) bf[j] = bp[(size_t)j * DIMM];
        acc = wmma_f16(af, bf, acc);
    }
    const int c = n0 + ln;
    const float bp = bias[c];
    #pragma unroll
    for (int i = 0; i < 8; ++i) {
        int mr = m0 + 8 * g + i;
        out[(size_t)mr * DIMM + c] = acc[i] + bp;
    }
}

// ---------------------------------------------------------------------------
// launcher
// ---------------------------------------------------------------------------
extern "C" void kernel_launch(void* const* d_in, const int* in_sizes, int n_in,
                              void* d_out, int out_size, void* d_ws, size_t ws_size,
                              hipStream_t stream) {
    (void)in_sizes; (void)n_in; (void)out_size; (void)ws_size;
    const float* x     = (const float*)d_in[0];
    const float* Wqkv  = (const float*)d_in[1];
    const float* Wproj = (const float*)d_in[2];
    const float* bproj = (const float*)d_in[3];
    float* out = (float*)d_out;

    // workspace carve-up (all sizes multiples of 256B)
    char* p = (char*)d_ws;
    _Float16* x16  = (_Float16*)p; p += (size_t)BB * TT * DIMM * 2;       // 8 MB
    _Float16* wq16 = (_Float16*)p; p += (size_t)DIMM * 2 * DIMM * 2;      // 1 MB
    _Float16* wp16 = (_Float16*)p; p += (size_t)DIMM * DIMM * 2;          // 0.5 MB
    _Float16* q16  = (_Float16*)p; p += (size_t)BB * HH * TT * HDD * 2;   // 8 MB
    _Float16* v16p = (_Float16*)p; p += (size_t)BB * HH * TT * HDD * 2;   // 8 MB
    float*    qsq  = (float*)p;    p += (size_t)BB * HH * TT * 4;         // 256 KB
    float*    scl  = (float*)p;    p += 256;
    unsigned* bmax = (unsigned*)p; p += 256;
    _Float16* oattn = x16;  // x16 is dead after the qkv GEMM -> reuse

    hipMemsetAsync(bmax, 0, BB * sizeof(unsigned), stream);

    k_convert_w<<<(DIMM * 2 * DIMM + 255) / 256, 256, 0, stream>>>(
        Wqkv, Wproj, wq16, wp16);
    k_xrow<<<BB * TT, 32, 0, stream>>>(x, x16, bmax);
    k_qkv_gemm<<<dim3((BB * TT) / 64, (2 * DIMM) / 32), 256, 0, stream>>>(
        x16, wq16, q16, v16p);
    k_qsq_scale<<<BB * HH, 256, 0, stream>>>(q16, qsq, bmax, scl);
    k_attn<<<dim3(TT / 128, BB * HH), 256, 0, stream>>>(
        q16, v16p, qsq, scl, oattn);
    k_proj_gemm<<<dim3((BB * TT) / 64, DIMM / 32), 256, 0, stream>>>(
        oattn, wp16, bproj, out);
}
